// AnyTop_42975442764379
// MI455X (gfx1250) — matrix operand order repaired and checked by reading, not compiled, not run on hardware
//
#include <hip/hip_runtime.h>
#include <hip/hip_bf16.h>
#include <math.h>

typedef __bf16 bf16_t;
typedef __attribute__((ext_vector_type(16))) __bf16 v16bf;
typedef __attribute__((ext_vector_type(8)))  float  v8f;
typedef __attribute__((ext_vector_type(4)))  unsigned int v4u;
typedef __attribute__((ext_vector_type(8)))  int v8i;
typedef __attribute__((ext_vector_type(4)))  int v4i;

#if defined(__has_builtin)
#  if __has_builtin(__builtin_amdgcn_tensor_load_to_lds)
#    define HAVE_TDM 1
#  endif
#endif
#ifndef HAVE_TDM
#  define HAVE_TDM 0
#endif
#if defined(__has_include)
#  if __has_include(<hip/amd_detail/amd_gfx1250_TDM.h>)
#    define TDM_6ARG 1
#  endif
#endif
#ifndef TDM_6ARG
#  define TDM_6ARG 0
#endif

namespace {
constexpr int kBS = 8, kNJ = 24, kNF = 13, kNFR = 160, kT = 161;
constexpr int kD = 256, kH = 4, kDH = 64, kFF = 1024, kL = 8, kT5 = 512;
constexpr int kM = kT * kBS * kNJ;     // 30912 tokens (divisible by 16)
constexpr int kKC = 256;               // K-chunk staged in LDS per GEMM block
constexpr float kScale = 0.125f;       // 1/sqrt(DH)
}

// ---------------- WMMA fragment loads (bf16, 16x16x32) ----------------
// A (16x32, MxK) per ISA: lane<16 -> M=lane, K = {0..7, 16..23}; lane>=16 -> K = {8..15, 24..31}
__device__ __forceinline__ v16bf load_a_frag(const bf16_t* p, int hi) {
  union { uint4 u[2]; v16bf v; } x;
  x.u[0] = *(const uint4*)(p + hi * 8);
  x.u[1] = *(const uint4*)(p + 16 + hi * 8);
  return x.v;
}
// B (32x16, KxN): lane%16 = N col; lanes 0-15 hold K = {0..7, 16..23}, lanes 16-31 K = {8..15, 24..31}.
__device__ __forceinline__ v16bf load_b_frag(const bf16_t* p, int hi) {
  union { uint4 u[2]; v16bf v; } x;
  x.u[0] = *(const uint4*)(p + hi * 8);
  x.u[1] = *(const uint4*)(p + 16 + hi * 8);
  return x.v;
}

__device__ __forceinline__ unsigned lds_addr_of(void* p) {
  return (unsigned)(unsigned long long)(__attribute__((address_space(3))) char*)p;
}

__device__ __forceinline__ float gelu_tanh(float x) {
  return 0.5f * x * (1.f + tanhf(0.7978845608028654f * (x + 0.044715f * x * x * x)));
}

// C[M,N] = act(A[M,K] x B[K,N] + bias); B supplied transposed as Bt[N,K].
// Block = 8 waves; each wave owns a 16x64 tile (4 accumulators). The 64-col
// B panel is staged K-chunk-wise into LDS (TDM tensor_load_to_lds when
// available) and shared by all 8 waves. Inner loop is software-pipelined:
// fragments for iteration kc+32 are issued before the WMMAs of iteration kc.
__global__ void __launch_bounds__(256)
gemm_wmma_bf16(const bf16_t* __restrict__ A,
               const bf16_t* __restrict__ Bt,
               const float* __restrict__ bias,
               float* __restrict__ Cf, bf16_t* __restrict__ Cb,
               int M, int N, int K, int act) {
  __shared__ __align__(16) bf16_t Bs[64][kKC + 8];  // +8 bf16 row pad (16B)
  const int wave = threadIdx.x >> 5;
  const int lane = threadIdx.x & 31;
  const int mt = blockIdx.x * 8 + wave;
  const bool active = (mt * 16 < M);
  const int mtc = active ? mt : 0;
  const int nbase = blockIdx.y * 64;
  const int r = lane & 15, hi = lane >> 4;
  const bf16_t* arow = A + (size_t)(mtc * 16 + r) * K;
  const v8f vzero = {0.f, 0.f, 0.f, 0.f, 0.f, 0.f, 0.f, 0.f};
  v8f acc[4];
#pragma unroll
  for (int n = 0; n < 4; ++n) acc[n] = vzero;

  for (int k0 = 0; k0 < K; k0 += kKC) {
    __syncthreads();   // previous chunk fully consumed
#if HAVE_TDM
    if (threadIdx.x < 32) {
      // TDM descriptor: 2D tile, 64 rows x kKC elements of 2B, row stride K,
      // LDS pad: 4 DWORDs (16B) every 128 DWORDs (512B = one row).
      unsigned long long ga =
          (unsigned long long)(const void*)(Bt + (size_t)nbase * K + k0);
      v4u g0;
      g0.x = 1u;                                   // count=1, user descriptor
      g0.y = lds_addr_of(&Bs[0][0]);               // LDS byte address
      g0.z = (unsigned)ga;                         // global addr [31:0]
      g0.w = (unsigned)((ga >> 32) & 0x1FFFFFFu) | (2u << 30);  // [56:32], type=2
      v8i g1;
      g1[0] = (int)((1u << 16)                     // data_size = 2 bytes
                    | (1u << 20)                   // pad_enable
                    | (6u << 22)                   // pad_interval = 128 DWORDs
                    | (3u << 25));                 // pad_amount   = 4 DWORDs
      g1[1] = (int)(((unsigned)K & 0xFFFFu) << 16);              // tensor_dim0 lo16
      g1[2] = (int)((((unsigned)K >> 16) & 0xFFFFu)              // tensor_dim0 hi16
                    | (((unsigned)N & 0xFFFFu) << 16));          // tensor_dim1 lo16
      g1[3] = (int)((((unsigned)N >> 16) & 0xFFFFu)              // tensor_dim1 hi16
                    | ((unsigned)kKC << 16));                    // tile_dim0
      g1[4] = 64;                                                // tile_dim1 (tile_dim2=0)
      g1[5] = (int)(unsigned)K;                                  // dim0_stride lo32
      g1[6] = 0;
      g1[7] = 0;
      v4i z4 = {0, 0, 0, 0};
#if TDM_6ARG
      v8i z8 = {0, 0, 0, 0, 0, 0, 0, 0};
      __builtin_amdgcn_tensor_load_to_lds(g0, g1, z4, z4, z8, 0);
#else
      __builtin_amdgcn_tensor_load_to_lds(g0, g1, z4, z4, 0);
#endif
#if __has_builtin(__builtin_amdgcn_s_wait_tensorcnt)
      __builtin_amdgcn_s_wait_tensorcnt(0);
#else
      asm volatile("s_wait_tensorcnt 0x0" ::: "memory");
#endif
    }
#else
    for (int u = threadIdx.x; u < 64 * (kKC / 8); u += 256) {
      int row = u >> 5;                 // kKC/8 = 32 x 16B units per row
      int k16 = (u & 31) * 8;
      *(uint4*)&Bs[row][k16] =
          *(const uint4*)(Bt + (size_t)(nbase + row) * K + k0 + k16);
    }
#endif
    __syncthreads();

    // Software-pipelined K loop: prefetch next A (global) + B (LDS) fragments
    // before consuming the current ones in the 4-WMMA chain.
    v16bf af = load_a_frag(arow + k0, hi);
    v16bf bcur[4];
#pragma unroll
    for (int n = 0; n < 4; ++n) bcur[n] = load_b_frag(&Bs[n * 16 + r][0], hi);
#pragma unroll
    for (int kc = 0; kc < kKC; kc += 32) {
      v16bf af_n = af;
      v16bf bnext[4];
      if (kc + 32 < kKC) {
        af_n = load_a_frag(arow + k0 + kc + 32, hi);
#pragma unroll
        for (int n = 0; n < 4; ++n)
          bnext[n] = load_b_frag(&Bs[n * 16 + r][kc + 32], hi);
      } else {
#pragma unroll
        for (int n = 0; n < 4; ++n) bnext[n] = bcur[n];
      }
#pragma unroll
      for (int n = 0; n < 4; ++n)
        acc[n] = __builtin_amdgcn_wmma_f32_16x16x32_bf16(false, af, false, bcur[n],
                                                         (short)0, acc[n], false, false);
      af = af_n;
#pragma unroll
      for (int n = 0; n < 4; ++n) bcur[n] = bnext[n];
    }
  }

  if (!active) return;
  // D layout: lane%16 = N col, VGPR j -> row j + 8*(lane/16)
#pragma unroll
  for (int n = 0; n < 4; ++n) {
    const int col = nbase + n * 16 + r;
    const float bv = bias ? bias[col] : 0.f;
#pragma unroll
    for (int j = 0; j < 8; ++j) {
      const int row = mt * 16 + hi * 8 + j;
      float v = acc[n][j] + bv;
      if (act == 1) v = gelu_tanh(v);
      if (Cf) Cf[(size_t)row * N + col] = v;
      if (Cb) Cb[(size_t)row * N + col] = (bf16_t)v;
    }
  }
}

// W[K,N] fp32 row-major -> Bt[N,K] bf16
__global__ void w_to_bt(const float* __restrict__ W, bf16_t* __restrict__ Bt, int K, int N) {
  int idx = blockIdx.x * 256 + threadIdx.x;
  if (idx >= K * N) return;
  int k = idx / N, n = idx - k * N;
  Bt[(size_t)n * K + k] = (bf16_t)W[idx];
}

// text[b,j,:] = joints_names_embs[b,j,:] @ W_text + b_text
__global__ void text_proj(const float* __restrict__ jne, const float* __restrict__ W_text,
                          const float* __restrict__ b_text, float* __restrict__ textb) {
  const int bj = blockIdx.x, d = threadIdx.x;
  const float* src = jne + (size_t)bj * kT5;
  float acc = b_text[d];
  for (int kk = 0; kk < kT5; ++kk) acc += src[kk] * W_text[kk * kD + d];
  textb[(size_t)bj * kD + d] = acc;
}

// Build h[t,b,j,:] = input proj + text + sin pos emb + timestep emb
__global__ void input_process(const float* __restrict__ x, const float* __restrict__ tpos,
                              const float* __restrict__ textb,
                              const float* __restrict__ W_root, const float* __restrict__ b_root,
                              const float* __restrict__ W_joint, const float* __restrict__ b_joint,
                              const float* __restrict__ W_troot, const float* __restrict__ b_troot,
                              const float* __restrict__ W_tjoint, const float* __restrict__ b_tjoint,
                              const int* __restrict__ timesteps, const int* __restrict__ crop,
                              float* __restrict__ Hf, bf16_t* __restrict__ Hb) {
  const int m = blockIdx.x, d = threadIdx.x;
  const int ti = m / (kBS * kNJ);
  const int rem = m - ti * kBS * kNJ;
  const int b = rem / kNJ;
  const int j = rem - b * kNJ;
  float acc;
  if (ti == 0) {
    const float* W  = (j == 0) ? W_troot : W_tjoint;
    const float* bv = (j == 0) ? b_troot : b_tjoint;
    const float* src = tpos + (size_t)(b * kNJ + j) * kNF;
    acc = bv[d];
    for (int f = 0; f < kNF; ++f) acc += src[f] * W[f * kD + d];
  } else {
    const float* W  = (j == 0) ? W_root : W_joint;
    const float* bv = (j == 0) ? b_root : b_joint;
    const float* src = x + (size_t)(b * kNJ + j) * kNF * kNFR + (ti - 1);
    acc = bv[d];
    for (int f = 0; f < kNF; ++f) acc += src[(size_t)f * kNFR] * W[f * kD + d];
  }
  acc += textb[(b * kNJ + j) * kD + d];
  const int a = (d < 128) ? d : d - 128;
  const float invfreq = powf(10000.f, (float)a * (1.f / 127.f));
  const float posv = (ti == 0) ? 0.f : (float)(ti + crop[0]);
  const float ph1 = posv / invfreq;
  acc += (d < 128) ? cosf(ph1) : sinf(ph1);
  const float ph2 = (float)timesteps[b] / invfreq;
  acc += (d < 128) ? cosf(ph2) : sinf(ph2);
  Hf[(size_t)m * kD + d] = acc;
  Hb[(size_t)m * kD + d] = (bf16_t)acc;
}

// Spatial attention: one block per (t*bs, head); 24-token softmax attention.
__global__ void spatial_attn(const bf16_t* __restrict__ Qb, const bf16_t* __restrict__ Kb,
                             const bf16_t* __restrict__ Vb, const int* __restrict__ jmask,
                             bf16_t* __restrict__ Ab) {
  const int tb = blockIdx.x;          // ti*kBS + b
  const int hh = blockIdx.y;
  const int b = tb % kBS;
  const int m0 = tb * kNJ;
  const int tid = threadIdx.x;
  __shared__ float q[kNJ][kDH + 1], k[kNJ][kDH + 1], v[kNJ][kDH + 1];
  __shared__ float s[kNJ][kNJ];
  for (int idx = tid; idx < kNJ * kDH; idx += 256) {
    int j = idx / kDH, c = idx - j * kDH;
    size_t g = (size_t)(m0 + j) * kD + hh * kDH + c;
    q[j][c] = (float)Qb[g]; k[j][c] = (float)Kb[g]; v[j][c] = (float)Vb[g];
  }
  __syncthreads();
  for (int idx = tid; idx < kNJ * kNJ; idx += 256) {
    int i = idx / kNJ, j = idx - i * kNJ;
    float acc = 0.f;
    for (int c = 0; c < kDH; ++c) acc += q[i][c] * k[j][c];
    int mk = jmask[b * 625 + (i + 1) * 25 + (j + 1)];
    s[i][j] = acc * kScale + (mk == 0 ? -1e9f : 0.f);
  }
  __syncthreads();
  if (tid < kNJ) {
    float mx = -3.4e38f;
    for (int j = 0; j < kNJ; ++j) mx = fmaxf(mx, s[tid][j]);
    float sum = 0.f;
    for (int j = 0; j < kNJ; ++j) { float e = __expf(s[tid][j] - mx); s[tid][j] = e; sum += e; }
    float inv = 1.f / sum;
    for (int j = 0; j < kNJ; ++j) s[tid][j] *= inv;
  }
  __syncthreads();
  for (int idx = tid; idx < kNJ * kDH; idx += 256) {
    int i = idx / kDH, c = idx - i * kDH;
    float acc = 0.f;
    for (int j = 0; j < kNJ; ++j) acc += s[i][j] * v[j][c];
    Ab[(size_t)(m0 + i) * kD + hh * kDH + c] = (bf16_t)acc;
  }
}

// Temporal attention: block per (b*nj, head, f-chunk); 161-token softmax attention.
__global__ void temporal_attn(const bf16_t* __restrict__ Qb, const bf16_t* __restrict__ Kb,
                              const bf16_t* __restrict__ Vb, const int* __restrict__ tmask,
                              bf16_t* __restrict__ Ab) {
  const int bj = blockIdx.x;          // b*kNJ + j
  const int hh = blockIdx.y;
  const int fbase = blockIdx.z * 32;
  const int b = bj / kNJ, jj = bj - b * kNJ;
  const int tid = threadIdx.x;
  __shared__ bf16_t Ks[kT][kDH], Vs[kT][kDH];   // 2 x 20.6 KB
  __shared__ float qrow[kDH];
  __shared__ float s[kT];
  __shared__ float red[256];
  __shared__ float part[4][kDH];
  for (int idx = tid; idx < kT * kDH; idx += 256) {
    int g = idx / kDH, c = idx - g * kDH;
    size_t gl = (size_t)((g * kBS + b) * kNJ + jj) * kD + hh * kDH + c;
    Ks[g][c] = Kb[gl];
    Vs[g][c] = Vb[gl];
  }
  __syncthreads();
  const int fend = (fbase + 32 < kT) ? fbase + 32 : kT;
  for (int f = fbase; f < fend; ++f) {
    const int mf = (f * kBS + b) * kNJ + jj;
    if (tid < kDH) qrow[tid] = (float)Qb[(size_t)mf * kD + hh * kDH + tid];
    __syncthreads();
    if (tid < kT) {
      float acc = 0.f;
      for (int c = 0; c < kDH; ++c) acc += qrow[c] * (float)Ks[tid][c];
      int mk = tmask[b * kT * kT + f * kT + tid];
      s[tid] = acc * kScale + (mk == 0 ? -1e9f : 0.f);
    }
    red[tid] = (tid < kT) ? s[tid] : -3.4e38f;
    __syncthreads();
    for (int o = 128; o > 0; o >>= 1) { if (tid < o) red[tid] = fmaxf(red[tid], red[tid + o]); __syncthreads(); }
    float mx = red[0];
    __syncthreads();
    float e = (tid < kT) ? __expf(s[tid] - mx) : 0.f;
    red[tid] = e;
    __syncthreads();
    for (int o = 128; o > 0; o >>= 1) { if (tid < o) red[tid] += red[tid + o]; __syncthreads(); }
    float inv = 1.f / red[0];
    __syncthreads();
    if (tid < kT) s[tid] = e * inv;
    __syncthreads();
    const int prt = tid >> 6, c = tid & 63;
    float acc = 0.f;
    for (int g = prt; g < kT; g += 4) acc += s[g] * (float)Vs[g][c];
    part[prt][c] = acc;
    __syncthreads();
    if (tid < kDH)
      Ab[(size_t)mf * kD + hh * kDH + tid] =
          (bf16_t)(part[0][tid] + part[1][tid] + part[2][tid] + part[3][tid]);
    __syncthreads();
  }
}

// Hf = LN(Hf + Cf) * g + b ; also write bf16 copy for next GEMM.
__global__ void ln_residual(float* __restrict__ Hf, const float* __restrict__ Cf,
                            const float* __restrict__ g, const float* __restrict__ bb,
                            bf16_t* __restrict__ Hb) {
  const int m = blockIdx.x, d = threadIdx.x;
  __shared__ float red[256];
  float val = Hf[(size_t)m * kD + d] + Cf[(size_t)m * kD + d];
  red[d] = val; __syncthreads();
  for (int o = 128; o > 0; o >>= 1) { if (d < o) red[d] += red[d + o]; __syncthreads(); }
  float mean = red[0] * (1.f / kD);
  __syncthreads();
  red[d] = val * val; __syncthreads();
  for (int o = 128; o > 0; o >>= 1) { if (d < o) red[d] += red[d + o]; __syncthreads(); }
  float var = red[0] * (1.f / kD) - mean * mean;
  float y = (val - mean) * rsqrtf(var + 1e-5f) * g[d] + bb[d];
  Hf[(size_t)m * kD + d] = y;
  Hb[(size_t)m * kD + d] = (bf16_t)y;
}

// out[b,j,f,fr] = h[fr+1,b,j,:] @ (j==0 ? W_rout : W_jout) + bias
__global__ void output_proc(const float* __restrict__ Hf,
                            const float* __restrict__ W_rout, const float* __restrict__ b_rout,
                            const float* __restrict__ W_jout, const float* __restrict__ b_jout,
                            float* __restrict__ out) {
  const int idx = blockIdx.x * 256 + threadIdx.x;
  const int ntot = kBS * kNJ * kNF * kNFR;
  if (idx >= ntot) return;
  const int fr = idx % kNFR;
  const int f  = (idx / kNFR) % kNF;
  const int j  = (idx / (kNFR * kNF)) % kNJ;
  const int b  = idx / (kNFR * kNF * kNJ);
  const int m  = ((fr + 1) * kBS + b) * kNJ + j;
  const float* W = (j == 0) ? W_rout : W_jout;
  float acc = (j == 0) ? b_rout[f] : b_jout[f];
  const float* hrow = Hf + (size_t)m * kD;
  for (int d = 0; d < kD; ++d) acc += hrow[d] * W[d * kNF + f];
  out[idx] = acc;
}

extern "C" void kernel_launch(void* const* d_in, const int* in_sizes, int n_in,
                              void* d_out, int out_size, void* d_ws, size_t ws_size,
                              hipStream_t stream) {
  (void)in_sizes; (void)n_in; (void)out_size; (void)ws_size;
  const float* x        = (const float*)d_in[0];
  const int*   tsteps   = (const int*)d_in[1];
  const int*   jmask    = (const int*)d_in[2];
  const int*   tmask    = (const int*)d_in[3];
  const float* tpos     = (const float*)d_in[4];
  const float* jne      = (const float*)d_in[5];
  const int*   crop     = (const int*)d_in[6];
  const float* W_root   = (const float*)d_in[7];  const float* b_root   = (const float*)d_in[8];
  const float* W_troot  = (const float*)d_in[9];  const float* b_troot  = (const float*)d_in[10];
  const float* W_joint  = (const float*)d_in[11]; const float* b_joint  = (const float*)d_in[12];
  const float* W_tjoint = (const float*)d_in[13]; const float* b_tjoint = (const float*)d_in[14];
  const float* W_text   = (const float*)d_in[15]; const float* b_text   = (const float*)d_in[16];
  const float* Wp[8]; const float* bp[8];          // q_s,k_s,v_s,o_s,q_t,k_t,v_t,o_t
  for (int i = 0; i < 8; ++i) { Wp[i] = (const float*)d_in[17 + 2*i]; bp[i] = (const float*)d_in[18 + 2*i]; }
  const float* W_ff1 = (const float*)d_in[33]; const float* b_ff1 = (const float*)d_in[34];
  const float* W_ff2 = (const float*)d_in[35]; const float* b_ff2 = (const float*)d_in[36];
  const float* ln_g[3] = {(const float*)d_in[37], (const float*)d_in[39], (const float*)d_in[41]};
  const float* ln_b[3] = {(const float*)d_in[38], (const float*)d_in[40], (const float*)d_in[42]};
  const float* W_rout = (const float*)d_in[43]; const float* b_rout = (const float*)d_in[44];
  const float* W_jout = (const float*)d_in[45]; const float* b_jout = (const float*)d_in[46];

  char* ws = (char*)d_ws; size_t off = 0;
  auto carve = [&](size_t bytes) -> char* {
    char* p = ws + off; off += (bytes + 255) & ~(size_t)255; return p;
  };
  float*  Hf    = (float*)carve((size_t)kM * kD * 4);
  float*  Cf    = (float*)carve((size_t)kM * kD * 4);
  float*  textb = (float*)carve((size_t)kBS * kNJ * kD * 4);
  // Qb..Ab carved contiguously (sizes are multiples of 256) so Fb[M,1024] aliases them.
  bf16_t* Qb   = (bf16_t*)carve((size_t)kM * kD * 2);
  bf16_t* Kbuf = (bf16_t*)carve((size_t)kM * kD * 2);
  bf16_t* Vbuf = (bf16_t*)carve((size_t)kM * kD * 2);
  bf16_t* Ab   = (bf16_t*)carve((size_t)kM * kD * 2);
  bf16_t* Fb   = Qb;
  bf16_t* Hb   = (bf16_t*)carve((size_t)kM * kD * 2);
  bf16_t* wt[kL][10];
  for (int l = 0; l < kL; ++l) {
    for (int i = 0; i < 8; ++i) wt[l][i] = (bf16_t*)carve((size_t)kD * kD * 2);
    wt[l][8] = (bf16_t*)carve((size_t)kFF * kD * 2);   // ff1^T [1024,256]
    wt[l][9] = (bf16_t*)carve((size_t)kD * kFF * 2);   // ff2^T [256,1024]
  }

  // ---- weight prep: transpose + bf16 ----
  for (int l = 0; l < kL; ++l) {
    for (int i = 0; i < 8; ++i)
      w_to_bt<<<dim3((kD * kD + 255) / 256), 256, 0, stream>>>(
          Wp[i] + (size_t)l * kD * kD, wt[l][i], kD, kD);
    w_to_bt<<<dim3((kD * kFF + 255) / 256), 256, 0, stream>>>(
        W_ff1 + (size_t)l * kD * kFF, wt[l][8], kD, kFF);
    w_to_bt<<<dim3((kFF * kD + 255) / 256), 256, 0, stream>>>(
        W_ff2 + (size_t)l * kFF * kD, wt[l][9], kFF, kD);
  }

  // ---- embed ----
  text_proj<<<dim3(kBS * kNJ), 256, 0, stream>>>(jne, W_text, b_text, textb);
  input_process<<<dim3(kM), 256, 0, stream>>>(x, tpos, textb, W_root, b_root, W_joint, b_joint,
                                              W_troot, b_troot, W_tjoint, b_tjoint,
                                              tsteps, crop, Hf, Hb);

  const dim3 blk(256);
  const int gx = (kM / 16 + 7) / 8;              // 1932 M-tiles, 8 per block -> 242
  const dim3 g256(gx, 4), g1024(gx, 16);
  const dim3 tgrid(kBS * kNJ, kH, (kT + 31) / 32);
  for (int l = 0; l < kL; ++l) {
    // spatial attention block
    gemm_wmma_bf16<<<g256, blk, 0, stream>>>(Hb, wt[l][0], bp[0] + l * kD, nullptr, Qb,   kM, kD, kD, 0);
    gemm_wmma_bf16<<<g256, blk, 0, stream>>>(Hb, wt[l][1], bp[1] + l * kD, nullptr, Kbuf, kM, kD, kD, 0);
    gemm_wmma_bf16<<<g256, blk, 0, stream>>>(Hb, wt[l][2], bp[2] + l * kD, nullptr, Vbuf, kM, kD, kD, 0);
    spatial_attn<<<dim3(kT * kBS, kH), 256, 0, stream>>>(Qb, Kbuf, Vbuf, jmask, Ab);
    gemm_wmma_bf16<<<g256, blk, 0, stream>>>(Ab, wt[l][3], bp[3] + l * kD, Cf, nullptr, kM, kD, kD, 0);
    ln_residual<<<dim3(kM), 256, 0, stream>>>(Hf, Cf, ln_g[0] + l * kD, ln_b[0] + l * kD, Hb);
    // temporal attention block
    gemm_wmma_bf16<<<g256, blk, 0, stream>>>(Hb, wt[l][4], bp[4] + l * kD, nullptr, Qb,   kM, kD, kD, 0);
    gemm_wmma_bf16<<<g256, blk, 0, stream>>>(Hb, wt[l][5], bp[5] + l * kD, nullptr, Kbuf, kM, kD, kD, 0);
    gemm_wmma_bf16<<<g256, blk, 0, stream>>>(Hb, wt[l][6], bp[6] + l * kD, nullptr, Vbuf, kM, kD, kD, 0);
    temporal_attn<<<tgrid, 256, 0, stream>>>(Qb, Kbuf, Vbuf, tmask, Ab);
    gemm_wmma_bf16<<<g256, blk, 0, stream>>>(Ab, wt[l][7], bp[7] + l * kD, Cf, nullptr, kM, kD, kD, 0);
    ln_residual<<<dim3(kM), 256, 0, stream>>>(Hf, Cf, ln_g[1] + l * kD, ln_b[1] + l * kD, Hb);
    // FFN
    gemm_wmma_bf16<<<g1024, blk, 0, stream>>>(Hb, wt[l][8], b_ff1 + (size_t)l * kFF, nullptr, Fb, kM, kFF, kD, 1);
    gemm_wmma_bf16<<<g256,  blk, 0, stream>>>(Fb, wt[l][9], b_ff2 + l * kD, Cf, nullptr, kM, kD, kFF, 0);
    ln_residual<<<dim3(kM), 256, 0, stream>>>(Hf, Cf, ln_g[2] + l * kD, ln_b[2] + l * kD, Hb);
  }

  const int nout = kBS * kNJ * kNF * kNFR;
  output_proc<<<dim3((nout + 255) / 256), 256, 0, stream>>>(Hf, W_rout, b_rout, W_jout, b_jout,
                                                            (float*)d_out);
}
